// DynamicSparseAttention_31722628448871
// MI455X (gfx1250) — matrix operand order repaired and checked by reading, not compiled, not run on hardware
//
#include <hip/hip_runtime.h>
#include <hip/hip_bf16.h>
#include <math.h>

// ---------------------------------------------------------------------------
// DynamicSparseAttention for MI455X (gfx1250, wave32, WMMA)
//   B=1, N=2048, C=1024, H=16, Dh=64, TOP_K=32
// ---------------------------------------------------------------------------

#define NSEQ   2048
#define CDIM   1024
#define HEADS  16
#define DH     64
#define TOPK   32
#define KDIM   1024          // reduction dim of every dense GEMM here
#define SC_STRIDE 2052       // padded LDS row stride (floats)

typedef __attribute__((ext_vector_type(16))) __bf16        v16bf;
typedef __attribute__((ext_vector_type(8)))  float         v8f;
typedef __attribute__((ext_vector_type(4)))  unsigned int  u32x4;

union FragB16 { v16bf v; u32x4 q[2]; };

__device__ __forceinline__ v8f wmma_bf16(v16bf a, v16bf b, v8f c) {
    // D = A(16x32 bf16) x B(32x16 bf16) + C(16x16 f32)
    return __builtin_amdgcn_wmma_f32_16x16x32_bf16(
        /*neg_a=*/false, a, /*neg_b=*/false, b,
        /*c_mod=*/(short)0, c, /*reuse_a=*/false, /*reuse_b=*/false);
}

// ---------------------------------------------------------------------------
// fp32 -> bf16 conversion (RNE)
// ---------------------------------------------------------------------------
__global__ __launch_bounds__(256) void cvt_bf16_kernel(const float* __restrict__ in,
                                                       __bf16* __restrict__ out, int n) {
    int i = blockIdx.x * 256 + threadIdx.x;
    if (i < n) out[i] = (__bf16)in[i];
}

// ---------------------------------------------------------------------------
// Register-blocked NT GEMM:
//   C[M=2048, N] = A[M,1024](bf16,K-contig) * W[N,1024](bf16,K-contig)^T
// Each wave computes a 64x64 macro-tile = 4x4 WMMA tiles (16 wmma per K-step
// against 8 fragment loads -> 32 FLOP/byte from cache). Block = 8 waves in a
// 2(M) x 4(N) arrangement -> 128x256 block tile.
// modes:
//   0: Q   : val=(acc+bias)*0.125 -> Qb   bf16 [H][N][Dh]
//   1: KV  : o<1024 -> Kb bf16 [H][N][Dh] ; o>=1024 -> Vf f32 [H][N][Dh]
//   2: proj: val=acc+bias -> outf f32 [M][1024]
// ---------------------------------------------------------------------------
__global__ __launch_bounds__(256) void gemm_nt_kernel(const __bf16* __restrict__ A,
                                                      const __bf16* __restrict__ W,
                                                      const float*  __restrict__ bias,
                                                      int mode,
                                                      __bf16* __restrict__ outb,
                                                      float*  __restrict__ outf) {
    const int wave  = threadIdx.x >> 5;
    const int lane  = threadIdx.x & 31;
    const int waveM = wave >> 2;                         // 0..1
    const int waveN = wave & 3;                          // 0..3
    const int mBase = blockIdx.y * 128 + waveM * 64;     // wave's 64 rows
    const int nBase = blockIdx.x * 256 + waveN * 64;     // wave's 64 cols
    const int l15   = lane & 15;
    const int kbA   = (lane >> 4) * 8;                   // A fragment sub-offset
    const int kbB   = (lane >> 4) * 8;                   // B fragment sub-offset

    const __bf16* aP[4];
    const __bf16* wP[4];
    #pragma unroll
    for (int i = 0; i < 4; ++i) aP[i] = A + (size_t)(mBase + i * 16 + l15) * KDIM;
    #pragma unroll
    for (int j = 0; j < 4; ++j) wP[j] = W + (size_t)(nBase + j * 16 + l15) * KDIM;

    v8f acc[4][4] = {};

    for (int k0 = 0; k0 < KDIM; k0 += 32) {
        FragB16 af[4], bf_[4];
        #pragma unroll
        for (int i = 0; i < 4; ++i) {
            af[i].q[0] = *(const u32x4*)(aP[i] + k0 + kbA);
            af[i].q[1] = *(const u32x4*)(aP[i] + k0 + kbA + 16);
        }
        #pragma unroll
        for (int j = 0; j < 4; ++j) {
            bf_[j].q[0] = *(const u32x4*)(wP[j] + k0 + kbB);
            bf_[j].q[1] = *(const u32x4*)(wP[j] + k0 + kbB + 16);
        }
        // speculative prefetch of next K-step (global_prefetch_b8; safe if OOB)
        __builtin_prefetch(aP[0] + k0 + 32, 0, 1);
        __builtin_prefetch(wP[0] + k0 + 32, 0, 1);
        #pragma unroll
        for (int i = 0; i < 4; ++i)
            #pragma unroll
            for (int j = 0; j < 4; ++j)
                acc[i][j] = wmma_bf16(af[i].v, bf_[j].v, acc[i][j]);
    }

    // epilogue: D layout per tile: lane&15 = col, (lane>>4)*8 + r = row
    #pragma unroll
    for (int j = 0; j < 4; ++j) {
        const int   o    = nBase + j * 16 + l15;
        const float bval = bias[o];
        #pragma unroll
        for (int i = 0; i < 4; ++i) {
            const int mB = mBase + i * 16 + (lane >> 4) * 8;
            #pragma unroll
            for (int r = 0; r < 8; ++r) {
                const int Mg = mB + r;
                float val = acc[i][j][r] + bval;
                if (mode == 0) {
                    val *= 0.125f;                       // fold softmax scale Dh^-0.5
                    const int h = o >> 6, d = o & 63;
                    outb[((size_t)h * NSEQ + Mg) * DH + d] = (__bf16)val;
                } else if (mode == 1) {
                    if (o < CDIM) {
                        const int h = o >> 6;
                        outb[((size_t)h * NSEQ + Mg) * DH + (o & 63)] = (__bf16)val;
                    } else {
                        const int oo = o - CDIM, h = oo >> 6;
                        outf[((size_t)h * NSEQ + Mg) * DH + (oo & 63)] = val;
                    }
                } else {
                    outf[(size_t)Mg * CDIM + o] = val;
                }
            }
        }
    }
}

// ---------------------------------------------------------------------------
// Per-head V row-sum:  Vsum[h][d] = sum_n Vf[h][n][d]
// ---------------------------------------------------------------------------
__global__ __launch_bounds__(64) void vsum_kernel(const float* __restrict__ Vf,
                                                  float* __restrict__ Vsum) {
    const int h = blockIdx.x, d = threadIdx.x;
    const float* p = Vf + (size_t)h * NSEQ * DH + d;
    float s = 0.f;
    for (int n = 0; n < NSEQ; ++n) s += p[(size_t)n * DH];
    Vsum[h * DH + d] = s;
}

// ---------------------------------------------------------------------------
// Attention: one block = one head x 16 query rows.
// Phase 1: 8 waves compute 16x2048 scores via WMMA into LDS (f32).
// Phase 2: each wave handles 2 rows: exact iterative top-32 extraction
//          (max value, ties -> lowest column, matching jax.lax.top_k),
//          masked-softmax closed form using per-head V row-sum, sparse P.V.
// ---------------------------------------------------------------------------
__global__ __launch_bounds__(256) void attn_kernel(const __bf16* __restrict__ Qb,
                                                   const __bf16* __restrict__ Kb,
                                                   const float*  __restrict__ Vf,
                                                   const float*  __restrict__ Vsum,
                                                   __bf16* __restrict__ aout) {
    extern __shared__ float sc[];                       // [16][SC_STRIDE]
    const int wave = threadIdx.x >> 5;
    const int lane = threadIdx.x & 31;
    const int h       = blockIdx.x >> 7;                // 128 row-tiles per head
    const int rowBase = (blockIdx.x & 127) * 16;

    // ---- Phase 1: scores = (Q*scale) . K^T ----
    const int m   = rowBase + (lane & 15);
    const int kbA = (lane >> 4) * 8;
    const int kbB = (lane >> 4) * 8;
    const __bf16* qRow = Qb + ((size_t)h * NSEQ + m) * DH;
    FragB16 a0, a1;
    a0.q[0] = *(const u32x4*)(qRow + kbA);
    a0.q[1] = *(const u32x4*)(qRow + kbA + 16);
    a1.q[0] = *(const u32x4*)(qRow + 32 + kbA);
    a1.q[1] = *(const u32x4*)(qRow + 32 + kbA + 16);

    for (int ct = 0; ct < 16; ++ct) {                   // 8 waves * 16 tiles = 2048 cols
        const int colBase = wave * 256 + ct * 16;
        const int n = colBase + (lane & 15);
        const __bf16* kRow = Kb + ((size_t)h * NSEQ + n) * DH;
        FragB16 b0, b1;
        b0.q[0] = *(const u32x4*)(kRow + kbB);
        b0.q[1] = *(const u32x4*)(kRow + kbB + 16);
        b1.q[0] = *(const u32x4*)(kRow + 32 + kbB);
        b1.q[1] = *(const u32x4*)(kRow + 32 + kbB + 16);
        v8f c = {};
        c = wmma_bf16(a0.v, b0.v, c);
        c = wmma_bf16(a1.v, b1.v, c);
        const int mloc = (lane >> 4) * 8;
        const int col  = colBase + (lane & 15);
        #pragma unroll
        for (int r = 0; r < 8; ++r)
            sc[(mloc + r) * SC_STRIDE + col] = c[r];
    }
    __syncthreads();

    // ---- Phase 2: top-32 + masked softmax + sparse PV ----
    for (int rr = 0; rr < 2; ++rr) {
        const int rloc = wave * 2 + rr;
        const int rowG = rowBase + rloc;

        float vals[64];                                 // lane owns cols j*32+lane
        #pragma unroll
        for (int j = 0; j < 64; ++j)
            vals[j] = sc[rloc * SC_STRIDE + j * 32 + lane];

        unsigned long long excl = 0ull;
        float selV = 0.f; int selC = 0; float topMax = 0.f;
        for (int t = 0; t < TOPK; ++t) {
            float bv = -INFINITY; int bj = 0;
            #pragma unroll
            for (int j = 0; j < 64; ++j) {
                const bool avail = ((excl >> j) & 1ull) == 0ull;
                if (avail && vals[j] > bv) { bv = vals[j]; bj = j; }
            }
            int bc = bj * 32 + lane;
            #pragma unroll
            for (int off = 16; off; off >>= 1) {        // wave32 butterfly argmax
                float ov = __shfl_xor(bv, off);
                int   oc = __shfl_xor(bc, off);
                if (ov > bv || (ov == bv && oc < bc)) { bv = ov; bc = oc; }
            }
            if ((bc & 31) == lane) excl |= (1ull << (bc >> 5));
            if (lane == t) { selV = bv; selC = bc; }
            if (t == 0) topMax = bv;
        }

        // softmax over (top-32 values, 2016 zeros)
        const float mprime = fmaxf(topMax, 0.f);
        const float w  = expf(selV - mprime);           // lane t holds weight t
        const float e0 = expf(-mprime);
        float wsum = w;
        #pragma unroll
        for (int off = 16; off; off >>= 1) wsum += __shfl_xor(wsum, off);
        const float denom = wsum + (float)(NSEQ - TOPK) * e0;

        // out[d] = (sum_sel w*v + e0*(Vsum - sum_sel v)) / denom ; lane -> dims 2l,2l+1
        float accx = 0.f, accy = 0.f, vsx = 0.f, vsy = 0.f;
        for (int t = 0; t < TOPK; ++t) {
            const float wt = __shfl(w, t);
            const int   ci = __shfl(selC, t);
            const float* vp = Vf + ((size_t)h * NSEQ + ci) * DH + 2 * lane;
            const float vx = vp[0], vy = vp[1];
            accx += wt * vx; accy += wt * vy;
            vsx  += vx;      vsy  += vy;
        }
        const float sx = Vsum[h * DH + 2 * lane];
        const float sy = Vsum[h * DH + 2 * lane + 1];
        const float ox = (accx + e0 * (sx - vsx)) / denom;
        const float oy = (accy + e0 * (sy - vsy)) / denom;
        __bf16* op = aout + (size_t)rowG * CDIM + h * DH + 2 * lane;
        op[0] = (__bf16)ox;
        op[1] = (__bf16)oy;
    }
}

// ---------------------------------------------------------------------------
// Host launcher
// ---------------------------------------------------------------------------
extern "C" void kernel_launch(void* const* d_in, const int* in_sizes, int n_in,
                              void* d_out, int out_size, void* d_ws, size_t ws_size,
                              hipStream_t stream) {
    const float* x     = (const float*)d_in[0];
    const float* Wq    = (const float*)d_in[1];
    const float* bq    = (const float*)d_in[2];
    const float* Wkv   = (const float*)d_in[3];
    const float* bkv   = (const float*)d_in[4];
    const float* Wproj = (const float*)d_in[5];
    const float* bproj = (const float*)d_in[6];
    float* out = (float*)d_out;

    // workspace layout (bf16 elements first, then f32; total ~32 MB)
    __bf16* xb   = (__bf16*)d_ws;                 // 2048*1024
    __bf16* wqb  = xb   + (size_t)NSEQ * CDIM;    // 1024*1024
    __bf16* wkvb = wqb  + (size_t)CDIM * CDIM;    // 2048*1024
    __bf16* wpb  = wkvb + (size_t)2 * CDIM * CDIM;// 1024*1024
    __bf16* Qb   = wpb  + (size_t)CDIM * CDIM;    // H*N*Dh
    __bf16* Kb   = Qb   + (size_t)HEADS * NSEQ * DH;
    __bf16* aout = Kb   + (size_t)HEADS * NSEQ * DH;
    float*  Vf   = (float*)(aout + (size_t)NSEQ * CDIM);
    float*  Vsum = Vf + (size_t)HEADS * NSEQ * DH;

    // 1) fp32 -> bf16 conversions
    cvt_bf16_kernel<<<(NSEQ * CDIM + 255) / 256, 256, 0, stream>>>(x, xb, NSEQ * CDIM);
    cvt_bf16_kernel<<<(CDIM * CDIM + 255) / 256, 256, 0, stream>>>(Wq, wqb, CDIM * CDIM);
    cvt_bf16_kernel<<<(2 * CDIM * CDIM + 255) / 256, 256, 0, stream>>>(Wkv, wkvb, 2 * CDIM * CDIM);
    cvt_bf16_kernel<<<(CDIM * CDIM + 255) / 256, 256, 0, stream>>>(Wproj, wpb, CDIM * CDIM);

    // 2) Q projection (scale folded), KV projection  (block tile 128x256)
    gemm_nt_kernel<<<dim3(CDIM / 256, NSEQ / 128), 256, 0, stream>>>(xb, wqb, bq, 0, Qb, nullptr);
    gemm_nt_kernel<<<dim3(2 * CDIM / 256, NSEQ / 128), 256, 0, stream>>>(xb, wkvb, bkv, 1, Kb, Vf);

    // 3) per-head V row-sums (for masked-softmax closed form)
    vsum_kernel<<<HEADS, DH, 0, stream>>>(Vf, Vsum);

    // 4) scores + top-32 + sparse softmax-V (128 KB dynamic LDS per block)
    const int lds_bytes = 16 * SC_STRIDE * (int)sizeof(float);
    (void)hipFuncSetAttribute((const void*)attn_kernel,
                              hipFuncAttributeMaxDynamicSharedMemorySize, lds_bytes);
    attn_kernel<<<HEADS * (NSEQ / 16), 256, lds_bytes, stream>>>(Qb, Kb, Vf, Vsum, aout);

    // 5) output projection -> d_out (fp32)
    gemm_nt_kernel<<<dim3(CDIM / 256, NSEQ / 128), 256, 0, stream>>>(aout, wpb, bproj, 2, nullptr, out);
}
